// FCInjection_75196287418385
// MI455X (gfx1250) — matrix operand order repaired and checked by reading, not compiled, not run on hardware
//
#include <hip/hip_runtime.h>
#include <hip/hip_bf16.h>
#include <math.h>

// ---------------------------------------------------------------------------
// Types for CDNA5 WMMA (wave32, v_wmma_f32_16x16x32_bf16)
// ---------------------------------------------------------------------------
typedef __bf16 bf16;
typedef __attribute__((ext_vector_type(8)))  __bf16 v8bf;
typedef __attribute__((ext_vector_type(16))) __bf16 v16bf;
typedef __attribute__((ext_vector_type(8)))  float  v8f;
typedef int v4i_ __attribute__((vector_size(16)));   // matches builtin proto

#if defined(__has_builtin)
#  if __has_builtin(__builtin_amdgcn_global_load_async_to_lds_b128)
#    define HAVE_ASYNC_G2L 1
#  endif
#  if __has_builtin(__builtin_amdgcn_permlane16)
#    define HAVE_PERMLANE16 1
#  endif
#endif

static __device__ inline v8f wmma_bf16f32(v16bf a, v16bf b, v8f c) {
  return __builtin_amdgcn_wmma_f32_16x16x32_bf16(false, a, false, b, (short)0, c,
                                                 false, false);
}
static __device__ inline v16bf cat16(v8bf lo, v8bf hi) {
  union { v16bf w; v8bf h[2]; } u;
  u.h[0] = lo; u.h[1] = hi;
  return u.w;
}
static __device__ inline v8f zero8() {
  v8f z;
#pragma unroll
  for (int i = 0; i < 8; ++i) z[i] = 0.0f;
  return z;
}

// 16-byte global -> LDS copy; async path (ASYNCcnt) when available.
// AS1 value of a generic global pointer is its 64-bit address; AS3 value of a
// generic LDS pointer is its low 32 bits (ISA 10.2), so integer reinterpret
// casts are numerically correct and avoid addrspace-cast legality issues.
static __device__ inline void cp16_g2l(const bf16* g, bf16* l) {
#ifdef HAVE_ASYNC_G2L
  __builtin_amdgcn_global_load_async_to_lds_b128(
      (__attribute__((address_space(1))) v4i_*)(unsigned long long)g,
      (__attribute__((address_space(3))) v4i_*)(unsigned int)(unsigned long long)l,
      0, 0);
#else
  *(v8bf*)l = *(const v8bf*)g;
#endif
}
#ifdef HAVE_ASYNC_G2L
#  define ASYNC_WAIT(imm) asm volatile("s_wait_asynccnt " #imm ::: "memory")
#else
#  define ASYNC_WAIT(imm)
#endif

// Butterfly max over the 16 lanes of a half-wave (rows of v_permlane16).
static __device__ inline float xormax16(float x) {
#ifdef HAVE_PERMLANE16
#  define PL16(y, s0, s1)                                                   \
    __int_as_float(__builtin_amdgcn_permlane16(__float_as_int(y),           \
        __float_as_int(y), (int)(s0), (int)(s1), false, false))
  x = fmaxf(x, PL16(x, 0x67452301, 0xefcdab89));  // xor 1
  x = fmaxf(x, PL16(x, 0x54761032, 0xdcfe98ba));  // xor 2
  x = fmaxf(x, PL16(x, 0x32107654, 0xba98fedc));  // xor 4
  x = fmaxf(x, PL16(x, 0xfedcba98, 0x76543210));  // xor 8
#  undef PL16
#else
#pragma unroll
  for (int m = 1; m < 16; m <<= 1) x = fmaxf(x, __shfl_xor(x, m, 32));
#endif
  return x;
}

// ---------------------------------------------------------------------------
// Setup kernels: fp32 -> bf16 converts (weights transposed to [N][K])
// ---------------------------------------------------------------------------
__global__ void cvt_f32_bf16(const float* __restrict__ src, bf16* __restrict__ dst,
                             int n) {
  int i = blockIdx.x * blockDim.x + threadIdx.x;
  if (i < n) dst[i] = (bf16)src[i];
}

// W [K,N] fp32 -> Wt [N,K] bf16
__global__ void cvt_transpose(const float* __restrict__ src, bf16* __restrict__ dst,
                              int K, int N) {
  int i = blockIdx.x * blockDim.x + threadIdx.x;
  if (i >= N * K) return;
  int n = i / K, k = i - n * K;
  dst[i] = (bf16)src[(size_t)k * N + n];
}

// Per-head weights [H=16, K, 64] fp32 -> [H*64, K] bf16 (row = h*64+d)
__global__ void repack_head_w_t(const float* __restrict__ src,
                                bf16* __restrict__ dst, int K) {
  int i = blockIdx.x * blockDim.x + threadIdx.x;
  if (i >= 1024 * K) return;
  int row = i / K, k = i - row * K;
  int h = row >> 6, d = row & 63;
  dst[i] = (bf16)src[((size_t)h * K + k) * 64 + d];
}

// ---------------------------------------------------------------------------
// Tiled WMMA GEMM:  out = act( A[M,K] @ Bt[N,K]^T (+bias) (+resid) )
// 256 threads = 8 waves (4x2); block tile 256x128, BK=32, wave tile 64x64.
// Double-buffered LDS staged with async global->LDS copies.
// OUT_MODE: 0 = f32 row-major, 1 = bf16 row-major, 2 = bf16 V-transposed
//           ([n,h][dh][t] layout for the attention V operand).
// ---------------------------------------------------------------------------
template <bool HAS_BIAS, bool RESID, bool TANH, int OUT_MODE>
__global__ __launch_bounds__(256)
void gemm_kernel(const bf16* __restrict__ A, const bf16* __restrict__ Bt,
                 const float* __restrict__ bias, const bf16* __restrict__ resid,
                 void* __restrict__ out, int M, int N, int K) {
  __shared__ bf16 lds[2][(256 + 128) * 32];
  const int tid = threadIdx.x;
  const int lane = tid & 31, wave = tid >> 5;
  const int waveM = wave >> 1, waveN = wave & 1;
  const int half = lane >> 4, l16 = lane & 15;
  const int m0 = blockIdx.x * 256, n0 = blockIdx.y * 128;

  v8f acc[4][4];
#pragma unroll
  for (int ms = 0; ms < 4; ++ms)
#pragma unroll
    for (int ns = 0; ns < 4; ++ns) acc[ms][ns] = zero8();

  auto stage = [&](int buf, int k0) {
    bf16* dA = &lds[buf][0];
    bf16* dB = &lds[buf][256 * 32];
#pragma unroll
    for (int j = 0; j < 4; ++j) {           // A: 256x32
      int flat = (tid + j * 256) * 8;
      int r = flat >> 5, c = flat & 31;
      cp16_g2l(A + (size_t)(m0 + r) * K + k0 + c, dA + flat);
    }
#pragma unroll
    for (int j = 0; j < 2; ++j) {           // Bt: 128x32 (already [n][k])
      int flat = (tid + j * 256) * 8;
      int r = flat >> 5, c = flat & 31;
      cp16_g2l(Bt + (size_t)(n0 + r) * K + k0 + c, dB + flat);
    }
  };

  const int nk = K >> 5;
  stage(0, 0);
  for (int i = 0; i < nk; ++i) {
    if (i + 1 < nk) { stage((i + 1) & 1, (i + 1) * 32); ASYNC_WAIT(6); }
    else            { ASYNC_WAIT(0); }
    __syncthreads();
    const bf16* sA = &lds[i & 1][0];
    const bf16* sB = &lds[i & 1][256 * 32];
    v16bf aF[4], bF[4];
#pragma unroll
    for (int ms = 0; ms < 4; ++ms) {
      const bf16* p = sA + (waveM * 64 + ms * 16 + l16) * 32;
      aF[ms] = cat16(*(const v8bf*)(p + half * 8),
                     *(const v8bf*)(p + 16 + half * 8));
    }
#pragma unroll
    for (int ns = 0; ns < 4; ++ns) {
      const bf16* p = sB + (waveN * 64 + ns * 16 + l16) * 32;
      bF[ns] = cat16(*(const v8bf*)(p + half * 8),
                     *(const v8bf*)(p + 16 + half * 8));
    }
#pragma unroll
    for (int ms = 0; ms < 4; ++ms)
#pragma unroll
      for (int ns = 0; ns < 4; ++ns)
        acc[ms][ns] = wmma_bf16f32(aF[ms], bF[ns], acc[ms][ns]);
    __syncthreads();
  }

  // Epilogue. C frag: row = v + 8*half, col = l16.
#pragma unroll
  for (int ms = 0; ms < 4; ++ms) {
#pragma unroll
    for (int ns = 0; ns < 4; ++ns) {
      int ng = n0 + waveN * 64 + ns * 16 + l16;
      float bv = 0.0f;
      if constexpr (HAS_BIAS) bv = bias[ng];
#pragma unroll
      for (int v = 0; v < 8; ++v) {
        int mg = m0 + waveM * 64 + ms * 16 + v + 8 * half;
        float x = acc[ms][ns][v] + bv;
        if constexpr (RESID) x += (float)resid[(size_t)mg * N + ng];
        if constexpr (TANH) x = tanhf(x);
        if constexpr (OUT_MODE == 0) {
          ((float*)out)[(size_t)mg * N + ng] = x;
        } else if constexpr (OUT_MODE == 1) {
          ((bf16*)out)[(size_t)mg * N + ng] = (bf16)x;
        } else {
          int nseq = mg >> 8, t = mg & 255, hh = ng >> 6, dh = ng & 63;
          ((bf16*)out)[((((size_t)nseq * 16 + hh) * 64 + dh) << 8) + t] = (bf16)x;
        }
      }
    }
  }
}

// ---------------------------------------------------------------------------
// Attention: one block per (n, h); 512 threads = 16 waves; wave w owns 16
// query rows. Full K tile (256x64) async-staged once; V streamed per 32-key
// chunk from the pre-transposed Vt with double buffering. Online softmax:
// row-max via permlane16 butterfly, row-sum via WMMA against ones.
// Q/K/Ctx layout: [n*256+s][h*64+dh]; Vt layout: [(n*16+h)][dh][t].
// ---------------------------------------------------------------------------
__global__ __launch_bounds__(512)
void attn_kernel(const bf16* __restrict__ Q, const bf16* __restrict__ Kb,
                 const bf16* __restrict__ Vt, bf16* __restrict__ Ctx) {
  __shared__ bf16 ldsK[256 * 64];      // [t][dh]
  __shared__ bf16 ldsV[2][64 * 32];    // [dh][t_local], double buffered
  __shared__ bf16 ldsP[16 * 512];      // per-wave prob scratch [s_local][t_local]
  const int nb = blockIdx.x >> 4, h = blockIdx.x & 15;
  const int tid = threadIdx.x, lane = tid & 31, wave = tid >> 5;
  const int half = lane >> 4, l16 = lane & 15;
  const size_t baseNH = (size_t)nb * 256 * 1024 + (size_t)h * 64;
  const bf16* vtBase = Vt + (((size_t)nb * 16 + h) << 14);  // *64*256

  // Stage K tile (async).
#pragma unroll
  for (int c = 0; c < 4; ++c) {
    int flat = (tid + c * 512) * 8;
    int t = flat >> 6, dh = flat & 63;
    cp16_g2l(Kb + baseNH + (size_t)t * 1024 + dh, ldsK + flat);
  }
  auto stageV = [&](int buf, int t0) {
    if (tid < 256) {
      int flat = tid * 8;                 // dh = flat>>5, tl = flat&31
      cp16_g2l(vtBase + (size_t)(flat >> 5) * 256 + t0 + (flat & 31),
               &ldsV[buf][flat]);
    }
  };

  // Q fragments: 16 rows x 64 dh -> 2 A-fragments.
  const int s0 = wave * 16;
  v16bf qF[2];
#pragma unroll
  for (int kc = 0; kc < 2; ++kc) {
    const bf16* qp = Q + baseNH + (size_t)(s0 + l16) * 1024 + kc * 32;
    qF[kc] = cat16(*(const v8bf*)(qp + half * 8),
                   *(const v8bf*)(qp + 16 + half * 8));
  }
  v16bf onesF;
#pragma unroll
  for (int i = 0; i < 16; ++i) onesF[i] = (bf16)1.0f;

  float runmax[8];
  v8f ctx[4], ctxS;
#pragma unroll
  for (int v = 0; v < 8; ++v) runmax[v] = -1e30f;
#pragma unroll
  for (int f = 0; f < 4; ++f) ctx[f] = zero8();
  ctxS = zero8();
  bf16* myP = ldsP + wave * 512;

  stageV(0, 0);
  for (int tc = 0; tc < 8; ++tc) {
    if (tc < 7) { stageV((tc + 1) & 1, (tc + 1) * 32); ASYNC_WAIT(1); }
    else        { ASYNC_WAIT(0); }
    __syncthreads();
    const int t0 = tc * 32;
    const bf16* sV = ldsV[tc & 1];

    // Scores: two 16x16 C frags over DH=64.
    v8f s[2];
    s[0] = zero8(); s[1] = zero8();
#pragma unroll
    for (int j = 0; j < 2; ++j) {
#pragma unroll
      for (int kc = 0; kc < 2; ++kc) {
        const bf16* kp = ldsK + (t0 + j * 16 + l16) * 64 + kc * 32 + half * 8;
        v16bf bF = cat16(*(const v8bf*)kp, *(const v8bf*)(kp + 16));
        s[j] = wmma_bf16f32(qF[kc], bF, s[j]);
      }
    }

    // Online softmax (row = v + 8*half).
    const float scale = 0.125f;  // 1/sqrt(64)
    float p0[8], p1[8];
#pragma unroll
    for (int v = 0; v < 8; ++v) {
      float a = s[0][v] * scale, b = s[1][v] * scale;
      float mx = xormax16(fmaxf(a, b));
      float mnew = fmaxf(runmax[v], mx);
      float alpha = __expf(runmax[v] - mnew);
      p0[v] = __expf(a - mnew);
      p1[v] = __expf(b - mnew);
      runmax[v] = mnew;
      ctxS[v] *= alpha;
#pragma unroll
      for (int f = 0; f < 4; ++f) ctx[f][v] *= alpha;
    }

    // Reshape probs C-frag -> A-frag through per-wave LDS scratch.
#pragma unroll
    for (int v = 0; v < 8; ++v) {
      int row = v + 8 * half;
      myP[row * 32 + l16]      = (bf16)p0[v];
      myP[row * 32 + 16 + l16] = (bf16)p1[v];
    }
    asm volatile("s_wait_dscnt 0" ::: "memory");
    v16bf pF = cat16(*(const v8bf*)(myP + l16 * 32 + half * 8),
                     *(const v8bf*)(myP + l16 * 32 + 16 + half * 8));

    // Row-sum via ones-WMMA (every column of ctxS holds the row sum).
    ctxS = wmma_bf16f32(pF, onesF, ctxS);
    // ctx += P(16x32) @ V(32x64).
#pragma unroll
    for (int f = 0; f < 4; ++f) {
      const bf16* vp = sV + (f * 16 + l16) * 32 + half * 8;
      v16bf bF = cat16(*(const v8bf*)vp, *(const v8bf*)(vp + 16));
      ctx[f] = wmma_bf16f32(pF, bF, ctx[f]);
    }
    __syncthreads();  // protect ldsV buffer reuse
  }

  // Normalize and store head-major ctx.
#pragma unroll
  for (int v = 0; v < 8; ++v) {
    float inv = 1.0f / ctxS[v];
    int row = s0 + v + 8 * half;
    bf16* op = Ctx + baseNH + (size_t)row * 1024;
#pragma unroll
    for (int f = 0; f < 4; ++f) op[f * 16 + l16] = (bf16)(ctx[f][v] * inv);
  }
}

// ---------------------------------------------------------------------------
// Host orchestration
// ---------------------------------------------------------------------------
extern "C" void kernel_launch(void* const* d_in, const int* in_sizes, int n_in,
                              void* d_out, int out_size, void* d_ws, size_t ws_size,
                              hipStream_t stream) {
  (void)in_sizes; (void)n_in; (void)out_size; (void)ws_size;
  const float* inp     = (const float*)d_in[0];
  const float* payload = (const float*)d_in[1];
  const float* W0 = (const float*)d_in[2];
  const float* b0 = (const float*)d_in[3];
  const float* W1 = (const float*)d_in[4];
  const float* b1 = (const float*)d_in[5];
  const float* W2 = (const float*)d_in[6];
  const float* b2 = (const float*)d_in[7];
  const float* Wo = (const float*)d_in[8];
  const float* bo = (const float*)d_in[9];
  const float* Wq = (const float*)d_in[10];
  const float* Wk = (const float*)d_in[11];
  const float* Wv = (const float*)d_in[12];
  const float* Wa = (const float*)d_in[13];
  float* out = (float*)d_out;

  constexpr int M = 16384, HID = 1024, IN = 768;

  char* ws = (char*)d_ws;
  size_t off = 0;
  auto take = [&](size_t elems) {
    bf16* p = (bf16*)(ws + off);
    off += elems * sizeof(bf16);
    return p;
  };
  bf16* bInp  = take((size_t)M * IN);
  bf16* bPay  = take((size_t)M * HID);
  bf16* bW0t  = take((size_t)IN * HID);
  bf16* bW1t  = take((size_t)HID * HID);
  bf16* bW2t  = take((size_t)HID * HID);
  bf16* bWot  = take((size_t)HID * HID);
  bf16* bWat  = take((size_t)HID * HID);
  bf16* bWqt  = take((size_t)HID * HID);
  bf16* bWkt  = take((size_t)HID * HID);
  bf16* bWvt  = take((size_t)HID * HID);
  bf16* Kb    = take((size_t)M * HID);
  bf16* Vtb   = take((size_t)M * HID);  // [(n*16+h)][dh][t]
  bf16* Xa    = take((size_t)M * HID);
  bf16* Xb    = take((size_t)M * HID);
  bf16* Qb    = take((size_t)M * HID);  // aliased: Q then Y (= x + inject)
  bf16* Cx    = take((size_t)M * HID);

  auto cvt = [&](const float* s, bf16* d, int n) {
    cvt_f32_bf16<<<(n + 255) / 256, 256, 0, stream>>>(s, d, n);
  };
  auto cvtT = [&](const float* s, bf16* d, int K, int N) {
    cvt_transpose<<<(K * N + 255) / 256, 256, 0, stream>>>(s, d, K, N);
  };
  cvt(inp,     bInp, M * IN);
  cvt(payload, bPay, M * HID);
  cvtT(W0, bW0t, IN,  HID);
  cvtT(W1, bW1t, HID, HID);
  cvtT(W2, bW2t, HID, HID);
  cvtT(Wo, bWot, HID, HID);
  cvtT(Wa, bWat, HID, HID);
  repack_head_w_t<<<(HID * HID + 255) / 256, 256, 0, stream>>>(Wq, bWqt, HID);
  repack_head_w_t<<<(HID * HID + 255) / 256, 256, 0, stream>>>(Wk, bWkt, HID);
  repack_head_w_t<<<(HID * HID + 255) / 256, 256, 0, stream>>>(Wv, bWvt, HID);

  dim3 blk(256);
  dim3 g(M / 256, HID / 128);
  dim3 ga(64 * 16);
  dim3 ba(512);

  // Layer 0: x = tanh(inp @ W0 + b0)
  gemm_kernel<true, false, true, 1><<<g, blk, 0, stream>>>(
      bInp, bW0t, b0, nullptr, (void*)Xa, M, HID, IN);
  // K/V depend only on payload -> compute once. V written pre-transposed.
  gemm_kernel<false, false, false, 1><<<g, blk, 0, stream>>>(
      bPay, bWkt, nullptr, nullptr, (void*)Kb, M, HID, HID);
  gemm_kernel<false, false, false, 2><<<g, blk, 0, stream>>>(
      bPay, bWvt, nullptr, nullptr, (void*)Vtb, M, HID, HID);

  // Inject 1: q = x@Wq; attn; y = x + ctx@Wa; x1 = y@W1 + b1
  gemm_kernel<false, false, false, 1><<<g, blk, 0, stream>>>(
      Xa, bWqt, nullptr, nullptr, (void*)Qb, M, HID, HID);
  attn_kernel<<<ga, ba, 0, stream>>>(Qb, Kb, Vtb, Cx);
  gemm_kernel<false, true, false, 1><<<g, blk, 0, stream>>>(
      Cx, bWat, nullptr, Xa, (void*)Qb, M, HID, HID);
  gemm_kernel<true, false, false, 1><<<g, blk, 0, stream>>>(
      Qb, bW1t, b1, nullptr, (void*)Xb, M, HID, HID);

  // Inject 2: q = x1@Wq; attn; y = x1 + ctx@Wa; x2 = y@W2 + b2
  gemm_kernel<false, false, false, 1><<<g, blk, 0, stream>>>(
      Xb, bWqt, nullptr, nullptr, (void*)Qb, M, HID, HID);
  attn_kernel<<<ga, ba, 0, stream>>>(Qb, Kb, Vtb, Cx);
  gemm_kernel<false, true, false, 1><<<g, blk, 0, stream>>>(
      Cx, bWat, nullptr, Xb, (void*)Qb, M, HID, HID);
  gemm_kernel<true, false, false, 1><<<g, blk, 0, stream>>>(
      Qb, bW2t, b2, nullptr, (void*)Xa, M, HID, HID);

  // Output: tanh(x2 @ Wo + bo), fp32
  gemm_kernel<true, false, true, 0><<<g, blk, 0, stream>>>(
      Xa, bWot, bo, nullptr, (void*)out, M, HID, HID);
}
